// ScaledDotProductAttention_26989574488108
// MI455X (gfx1250) — matrix operand order repaired and checked by reading, not compiled, not run on hardware
//
#include <hip/hip_runtime.h>
#include <stdint.h>

// ---------------------------------------------------------------------------
// Flash attention (B=2,H=16,S=2048,D=64) fp32 + bool mask, no 1/sqrt(d).
// bf16 WMMA (v_wmma_f32_16x16x32_bf16) with f32 accumulation, online softmax.
// - one-shot K/V -> bf16 (V transposed) workspace pass
// - main kernel double-buffers K/V tiles via global_load_async_to_lds_b128
//   (ASYNCcnt-tracked), overlapped with compute
// - softmax row-max via DPP row_ror rotate-reduction (pure VALU, no LDS)
// - softmax row-sum via P @ ones WMMA (replaces 32 ds_bpermute round trips)
// Mask (134MB, streamed once, NT) is the traffic floor ~6us @ 23.3TB/s.
// ---------------------------------------------------------------------------

typedef __attribute__((ext_vector_type(16))) __bf16    v16bf;
typedef __attribute__((ext_vector_type(8)))  float     v8f;
typedef __attribute__((ext_vector_type(4)))  float     f32x4;
typedef __attribute__((ext_vector_type(4)))  uint32_t  u32x4;
typedef __attribute__((ext_vector_type(2)))  uint32_t  u32x2;
typedef __attribute__((ext_vector_type(4)))  int       i32x4;

#define S_LEN   2048
#define DHEAD   64
#define BLK_K   64              // keys per LDS block
#define WAVES   8
#define BLK_Q   (16 * WAVES)    // 128 queries per workgroup
#define NBH     32              // B*H
#define NEGBIG  (-1000000000.0f)

// LDS row strides in bf16 elements (72*2 = 144B: 16B-aligned, bank-staggered)
#define KROW 72
#define VROW 72
#define PROW 72
#define KBUF (BLK_K * KROW)     // one K/V LDS buffer, elements
#define KBUFB (KBUF * 2)        // bytes (9216)

union V16U {
  v16bf    v;
  u32x4    q[2];
  uint32_t w[8];
  __bf16   h[16];
};

// 16-bit A-matrix 16x32 element->K mapping (contiguous, matches B loads):
// element e holds K = 16*half + e
__device__ __forceinline__ int a_kk(int e, int half) {
  return 16 * half + e;
}

// B-operand of all-ones bf16 (for P @ ones row-sum WMMA)
__device__ __forceinline__ v16bf ones16() {
  V16U o;
#pragma unroll
  for (int j = 0; j < 8; ++j) o.w[j] = 0x3F803F80u;  // bf16 1.0 pair
  return o.v;
}

// max across the 16-lane DPP row via rotate-reduction (all lanes get result)
__device__ __forceinline__ float rowmax16(float x) {
#if __has_builtin(__builtin_amdgcn_update_dpp)
  x = fmaxf(x, __int_as_float(__builtin_amdgcn_update_dpp(
                   0, __float_as_int(x), 0x128, 0xf, 0xf, true)));  // row_ror:8
  x = fmaxf(x, __int_as_float(__builtin_amdgcn_update_dpp(
                   0, __float_as_int(x), 0x124, 0xf, 0xf, true)));  // row_ror:4
  x = fmaxf(x, __int_as_float(__builtin_amdgcn_update_dpp(
                   0, __float_as_int(x), 0x122, 0xf, 0xf, true)));  // row_ror:2
  x = fmaxf(x, __int_as_float(__builtin_amdgcn_update_dpp(
                   0, __float_as_int(x), 0x121, 0xf, 0xf, true)));  // row_ror:1
#else
  x = fmaxf(x, __shfl_xor(x, 8, 32));
  x = fmaxf(x, __shfl_xor(x, 4, 32));
  x = fmaxf(x, __shfl_xor(x, 2, 32));
  x = fmaxf(x, __shfl_xor(x, 1, 32));
#endif
  return x;
}

// --------------------------- async copy helpers ----------------------------
// Builtin signature (probe-verified): (int4 AS1* src, int4 AS3* dst,
// Ii offset, Ii cpol) -- llvm.amdgcn.global.load.async.to.lds.b128.
typedef __attribute__((address_space(1))) i32x4 gi32x4_t;
typedef __attribute__((address_space(3))) i32x4 li32x4_t;

__device__ __forceinline__ void async_copy16(const void* g, void* l) {
#if __has_builtin(__builtin_amdgcn_global_load_async_to_lds_b128)
  __builtin_amdgcn_global_load_async_to_lds_b128(
      (gi32x4_t*)(uintptr_t)g, (li32x4_t*)(uint32_t)(uintptr_t)l, 0, 0);
#else
  asm volatile("global_load_async_to_lds_b128 %0, %1, off"
               :: "v"((uint32_t)(uintptr_t)l), "v"((uint64_t)(uintptr_t)g)
               : "memory");
#endif
}
__device__ __forceinline__ void wait_async0() {
#if __has_builtin(__builtin_amdgcn_s_wait_asynccnt)
  __builtin_amdgcn_s_wait_asynccnt(0);
#else
  asm volatile("s_wait_asynccnt 0x0" ::: "memory");
#endif
}

// ---------------------------------------------------------------------------
// Prep pass: K -> bf16 [bh][s][d]; V -> bf16 transposed [bh][d][s].
// ---------------------------------------------------------------------------
__global__ __launch_bounds__(256) void cvt_kv_kernel(
    const float* __restrict__ K, const float* __restrict__ V,
    __bf16* __restrict__ Kb, __bf16* __restrict__ VTb)
{
  __shared__ __bf16 sT[DHEAD * 72];     // [d][s] tile
  const int bh = blockIdx.x >> 5;
  const int sb = (blockIdx.x & 31) << 6;
  const int tid = threadIdx.x;
  const float* Kg = K + ((size_t)bh * S_LEN + sb) * DHEAD;
  const float* Vg = V + ((size_t)bh * S_LEN + sb) * DHEAD;
  __bf16* Ko = Kb + ((size_t)bh * S_LEN + sb) * DHEAD;
#pragma unroll
  for (int rep = 0; rep < 4; ++rep) {
    int linear = rep * 256 + tid;       // 1024 x 4 floats = 64x64
    int row = linear >> 4;
    int c4  = (linear & 15) << 2;
    f32x4 kf = *(const f32x4*)(Kg + row * DHEAD + c4);
    f32x4 vf = *(const f32x4*)(Vg + row * DHEAD + c4);
    union { __bf16 b[4]; u32x2 u; } kp;
    kp.b[0] = (__bf16)kf.x; kp.b[1] = (__bf16)kf.y;
    kp.b[2] = (__bf16)kf.z; kp.b[3] = (__bf16)kf.w;
    *(u32x2*)(Ko + row * DHEAD + c4) = kp.u;
    sT[(c4 + 0) * 72 + row] = (__bf16)vf.x;
    sT[(c4 + 1) * 72 + row] = (__bf16)vf.y;
    sT[(c4 + 2) * 72 + row] = (__bf16)vf.z;
    sT[(c4 + 3) * 72 + row] = (__bf16)vf.w;
  }
  __syncthreads();
  {
    int d   = tid >> 2;                 // 64 rows x 4 threads x 16 halves
    int off = (tid & 3) << 4;
    __bf16* dst = VTb + ((size_t)bh * DHEAD + d) * S_LEN + sb + off;
    const __bf16* src = sT + d * 72 + off;
    *(u32x4*)(dst)     = *(const u32x4*)(src);
    *(u32x4*)(dst + 8) = *(const u32x4*)(src + 8);
  }
}

// ---------------------------------------------------------------------------
// Main kernel: async double-buffered bf16 K/V tiles.
// ---------------------------------------------------------------------------
__device__ __forceinline__ void issue_block(
    const __bf16* Kg, const __bf16* VTg, __bf16* sK, __bf16* sVT,
    int kb, int bufsel, int wave, int lane)
{
#pragma unroll
  for (int c = 0; c < 2; ++c) {
    int i = wave + c * 8;               // chunk 0..15 of the 8KB tile
    int o = i * 512 + lane * 16;        // byte offset in tile
    int row = o >> 7, inrow = o & 127;  // 128B per (key|d) row
    async_copy16((const char*)Kg + (size_t)(kb + row) * 128 + inrow,
                 (char*)sK + bufsel * KBUFB + row * 144 + inrow);
    async_copy16((const char*)VTg + (size_t)row * (S_LEN * 2) + (size_t)kb * 2 + inrow,
                 (char*)sVT + bufsel * KBUFB + row * 144 + inrow);
  }
}

__global__ __launch_bounds__(256) void fa_async_kernel(
    const float* __restrict__ Q, const __bf16* __restrict__ Kb,
    const __bf16* __restrict__ VTb, const unsigned char* __restrict__ Msk,
    float* __restrict__ O)
{
  __shared__ __bf16 sK [2 * KBUF];            // double-buffered [key][d]
  __shared__ __bf16 sVT[2 * KBUF];            // double-buffered [d][key]
  __shared__ __bf16 sP [WAVES * 16 * PROW];   // per-wave P scratch

  const int tid  = threadIdx.x;
  const int wave = tid >> 5;
  const int lane = tid & 31;
  const int half = lane >> 4;
  const int ln   = lane & 15;

  const int bh = blockIdx.y;
  const int q0 = blockIdx.x * BLK_Q + wave * 16;

  const float*  Qg  = Q   + (size_t)bh * S_LEN * DHEAD;
  const __bf16* Kg  = Kb  + (size_t)bh * S_LEN * DHEAD;
  const __bf16* VTg = VTb + (size_t)bh * DHEAD * S_LEN;
  float*        Og  = O   + (size_t)bh * S_LEN * DHEAD;
  const unsigned char* Mg = Msk + (size_t)bh * (size_t)S_LEN * (size_t)S_LEN;

  const v16bf vones = ones16();

  // ---- Q tile (16 x 64) -> two A-layout bf16 register matrices ----
  V16U qa0, qa1;
  {
    const float* qrow = Qg + (size_t)(q0 + ln) * DHEAD;
#pragma unroll
    for (int e = 0; e < 16; ++e) {
      int kk = a_kk(e, half);
      qa0.h[e] = (__bf16)qrow[kk];
      qa1.h[e] = (__bf16)qrow[32 + kk];
    }
  }

  float m_i[8], l_i[8];
  v8f   accO[4];
#pragma unroll
  for (int i = 0; i < 8; ++i) { m_i[i] = -3.0e38f; l_i[i] = 0.0f; }
#pragma unroll
  for (int dt = 0; dt < 4; ++dt) accO[dt] = (v8f)(0.0f);

  __bf16* sPw = sP + wave * 16 * PROW;

  // prologue: fetch block 0
  issue_block(Kg, VTg, sK, sVT, 0, 0, wave, lane);
  wait_async0();
  __syncthreads();

  for (int kb = 0, n = 0; kb < S_LEN; kb += BLK_K, ++n) {
    const int cur = n & 1;
    if (kb + BLK_K < S_LEN)   // overlap next block's copy with this compute
      issue_block(Kg, VTg, sK, sVT, kb + BLK_K, cur ^ 1, wave, lane);

    const __bf16* sKc = sK  + cur * KBUF;
    const __bf16* sVc = sVT + cur * KBUF;

    // ---- S = Q @ K^T : batch all B loads, then 4 independent WMMA chains ----
    V16U bq[8];
#pragma unroll
    for (int t = 0; t < 4; ++t) {
      const __bf16* krow = sKc + (t * 16 + ln) * KROW + half * 16;
      bq[t].q[0]     = *(const u32x4*)(krow);
      bq[t].q[1]     = *(const u32x4*)(krow + 8);
      bq[4 + t].q[0] = *(const u32x4*)(krow + 32);
      bq[4 + t].q[1] = *(const u32x4*)(krow + 40);
    }
    v8f sreg[4];
#pragma unroll
    for (int t = 0; t < 4; ++t)
      sreg[t] = __builtin_amdgcn_wmma_f32_16x16x32_bf16(
          false, qa0.v, false, bq[t].v, (short)0, (v8f)(0.0f), false, false);
#pragma unroll
    for (int t = 0; t < 4; ++t)
      sreg[t] = __builtin_amdgcn_wmma_f32_16x16x32_bf16(
          false, qa1.v, false, bq[4 + t].v, (short)0, sreg[t], false, false);

    // ---- mask (non-temporal: 134MB streamed once, keep L2 for K/V) ----
#pragma unroll
    for (int t = 0; t < 4; ++t)
#pragma unroll
      for (int i = 0; i < 8; ++i) {
        unsigned char mv = __builtin_nontemporal_load(
            Mg + (size_t)(q0 + i + 8 * half) * S_LEN + (kb + t * 16 + ln));
        sreg[t][i] = mv ? NEGBIG : sreg[t][i];
      }

    // ---- online softmax: row max via DPP rotate-reduce (no LDS) ----
    float mnew[8], alpha[8];
#pragma unroll
    for (int i = 0; i < 8; ++i) {
      float mx = fmaxf(fmaxf(sreg[0][i], sreg[1][i]),
                       fmaxf(sreg[2][i], sreg[3][i]));
      mx = rowmax16(mx);
      mnew[i]  = fmaxf(m_i[i], mx);
      alpha[i] = __expf(m_i[i] - mnew[i]);
      m_i[i]   = mnew[i];
    }
#pragma unroll
    for (int dt = 0; dt < 4; ++dt)
#pragma unroll
      for (int i = 0; i < 8; ++i) accO[dt][i] = accO[dt][i] * alpha[i];

#pragma unroll
    for (int t = 0; t < 4; ++t)
#pragma unroll
      for (int i = 0; i < 8; ++i) {
        float p = __expf(sreg[t][i] - mnew[i]);
        sPw[(i + 8 * half) * PROW + (t * 16 + ln)] = (__bf16)p;  // C -> LDS
      }

    // ---- O += P @ V, and row-sum l += P @ ones (WMMA, no butterflies) ----
    v8f accL = (v8f)(0.0f);
#pragma unroll
    for (int c = 0; c < 2; ++c) {
      V16U pa, vb[4];
      const __bf16* prow = sPw + ln * PROW + c * 32;
#pragma unroll
      for (int j = 0; j < 8; ++j) {
        int kk = 16 * half + (j << 1);
        pa.w[j] = *(const uint32_t*)(prow + kk);
      }
#pragma unroll
      for (int dt = 0; dt < 4; ++dt) {
        const __bf16* vrow = sVc + (dt * 16 + ln) * VROW + c * 32 + half * 16;
        vb[dt].q[0] = *(const u32x4*)(vrow);
        vb[dt].q[1] = *(const u32x4*)(vrow + 8);
      }
      accL = __builtin_amdgcn_wmma_f32_16x16x32_bf16(
          false, pa.v, false, vones, (short)0, accL, false, false);
#pragma unroll
      for (int dt = 0; dt < 4; ++dt)
        accO[dt] = __builtin_amdgcn_wmma_f32_16x16x32_bf16(
            false, pa.v, false, vb[dt].v, (short)0, accO[dt], false, false);
    }
#pragma unroll
    for (int i = 0; i < 8; ++i) l_i[i] = l_i[i] * alpha[i] + accL[i];

    wait_async0();      // next block's tiles resident
    __syncthreads();    // everyone done reading current buffers
  }

  // ---- epilogue: O = acc / l ----
  float inv[8];
#pragma unroll
  for (int i = 0; i < 8; ++i) inv[i] = 1.0f / l_i[i];
#pragma unroll
  for (int dt = 0; dt < 4; ++dt)
#pragma unroll
    for (int i = 0; i < 8; ++i)
      __builtin_nontemporal_store(
          accO[dt][i] * inv[i],
          Og + (size_t)(q0 + i + 8 * half) * DHEAD + dt * 16 + ln);
}

// ---------------------------------------------------------------------------
// Fallback (no workspace): self-contained kernel, fp32->bf16 in-loop.
// ---------------------------------------------------------------------------
__global__ __launch_bounds__(256) void fa_fallback_kernel(
    const float* __restrict__ Q, const float* __restrict__ K,
    const float* __restrict__ V, const unsigned char* __restrict__ Msk,
    float* __restrict__ O)
{
  __shared__ __bf16 sK [BLK_K * KROW];
  __shared__ __bf16 sVT[DHEAD * VROW];
  __shared__ __bf16 sP [WAVES * 16 * PROW];

  const int tid  = threadIdx.x;
  const int wave = tid >> 5;
  const int lane = tid & 31;
  const int half = lane >> 4;
  const int ln   = lane & 15;
  const int bh = blockIdx.y;
  const int q0 = blockIdx.x * BLK_Q + wave * 16;

  const size_t base = (size_t)bh * S_LEN * DHEAD;
  const float* Qg = Q + base;
  const float* Kg = K + base;
  const float* Vg = V + base;
  float*       Og = O + base;
  const unsigned char* Mg = Msk + (size_t)bh * (size_t)S_LEN * (size_t)S_LEN;

  const v16bf vones = ones16();

  V16U qa0, qa1;
  {
    const float* qrow = Qg + (size_t)(q0 + ln) * DHEAD;
#pragma unroll
    for (int e = 0; e < 16; ++e) {
      int kk = a_kk(e, half);
      qa0.h[e] = (__bf16)qrow[kk];
      qa1.h[e] = (__bf16)qrow[32 + kk];
    }
  }
  float m_i[8], l_i[8];
  v8f accO[4];
#pragma unroll
  for (int i = 0; i < 8; ++i) { m_i[i] = -3.0e38f; l_i[i] = 0.0f; }
#pragma unroll
  for (int dt = 0; dt < 4; ++dt) accO[dt] = (v8f)(0.0f);
  __bf16* sPw = sP + wave * 16 * PROW;

  for (int kb = 0; kb < S_LEN; kb += BLK_K) {
    __syncthreads();
#pragma unroll
    for (int rep = 0; rep < 4; ++rep) {
      int linear = rep * 256 + tid;
      int row = linear >> 4;
      int c4  = (linear & 15) << 2;
      f32x4 kf = *(const f32x4*)(Kg + (size_t)(kb + row) * DHEAD + c4);
      f32x4 vf = *(const f32x4*)(Vg + (size_t)(kb + row) * DHEAD + c4);
      union { __bf16 b[4]; u32x2 u; } kp;
      kp.b[0] = (__bf16)kf.x; kp.b[1] = (__bf16)kf.y;
      kp.b[2] = (__bf16)kf.z; kp.b[3] = (__bf16)kf.w;
      *(u32x2*)(sK + row * KROW + c4) = kp.u;
      sVT[(c4 + 0) * VROW + row] = (__bf16)vf.x;
      sVT[(c4 + 1) * VROW + row] = (__bf16)vf.y;
      sVT[(c4 + 2) * VROW + row] = (__bf16)vf.z;
      sVT[(c4 + 3) * VROW + row] = (__bf16)vf.w;
    }
    __syncthreads();

    V16U bq[8];
#pragma unroll
    for (int t = 0; t < 4; ++t) {
      const __bf16* krow = sK + (t * 16 + ln) * KROW + half * 16;
      bq[t].q[0]     = *(const u32x4*)(krow);
      bq[t].q[1]     = *(const u32x4*)(krow + 8);
      bq[4 + t].q[0] = *(const u32x4*)(krow + 32);
      bq[4 + t].q[1] = *(const u32x4*)(krow + 40);
    }
    v8f sreg[4];
#pragma unroll
    for (int t = 0; t < 4; ++t)
      sreg[t] = __builtin_amdgcn_wmma_f32_16x16x32_bf16(
          false, qa0.v, false, bq[t].v, (short)0, (v8f)(0.0f), false, false);
#pragma unroll
    for (int t = 0; t < 4; ++t)
      sreg[t] = __builtin_amdgcn_wmma_f32_16x16x32_bf16(
          false, qa1.v, false, bq[4 + t].v, (short)0, sreg[t], false, false);
#pragma unroll
    for (int t = 0; t < 4; ++t)
#pragma unroll
      for (int i = 0; i < 8; ++i) {
        unsigned char mv = __builtin_nontemporal_load(
            Mg + (size_t)(q0 + i + 8 * half) * S_LEN + (kb + t * 16 + ln));
        sreg[t][i] = mv ? NEGBIG : sreg[t][i];
      }
    float mnew[8], alpha[8];
#pragma unroll
    for (int i = 0; i < 8; ++i) {
      float mx = fmaxf(fmaxf(sreg[0][i], sreg[1][i]),
                       fmaxf(sreg[2][i], sreg[3][i]));
      mx = rowmax16(mx);
      mnew[i]  = fmaxf(m_i[i], mx);
      alpha[i] = __expf(m_i[i] - mnew[i]);
      m_i[i]   = mnew[i];
    }
#pragma unroll
    for (int dt = 0; dt < 4; ++dt)
#pragma unroll
      for (int i = 0; i < 8; ++i) accO[dt][i] = accO[dt][i] * alpha[i];
#pragma unroll
    for (int t = 0; t < 4; ++t)
#pragma unroll
      for (int i = 0; i < 8; ++i) {
        float p = __expf(sreg[t][i] - mnew[i]);
        sPw[(i + 8 * half) * PROW + (t * 16 + ln)] = (__bf16)p;
      }
    v8f accL = (v8f)(0.0f);
#pragma unroll
    for (int c = 0; c < 2; ++c) {
      V16U pa, vb[4];
      const __bf16* prow = sPw + ln * PROW + c * 32;
#pragma unroll
      for (int j = 0; j < 8; ++j) {
        int kk = 16 * half + (j << 1);
        pa.w[j] = *(const uint32_t*)(prow + kk);
      }
#pragma unroll
      for (int dt = 0; dt < 4; ++dt) {
        const __bf16* vrow = sVT + (dt * 16 + ln) * VROW + c * 32 + half * 16;
        vb[dt].q[0] = *(const u32x4*)(vrow);
        vb[dt].q[1] = *(const u32x4*)(vrow + 8);
      }
      accL = __builtin_amdgcn_wmma_f32_16x16x32_bf16(
          false, pa.v, false, vones, (short)0, accL, false, false);
#pragma unroll
      for (int dt = 0; dt < 4; ++dt)
        accO[dt] = __builtin_amdgcn_wmma_f32_16x16x32_bf16(
            false, pa.v, false, vb[dt].v, (short)0, accO[dt], false, false);
    }
#pragma unroll
    for (int i = 0; i < 8; ++i) l_i[i] = l_i[i] * alpha[i] + accL[i];
  }
  float inv[8];
#pragma unroll
  for (int i = 0; i < 8; ++i) inv[i] = 1.0f / l_i[i];
#pragma unroll
  for (int dt = 0; dt < 4; ++dt)
#pragma unroll
    for (int i = 0; i < 8; ++i)
      Og[(size_t)(q0 + i + 8 * half) * DHEAD + dt * 16 + ln] =
          accO[dt][i] * inv[i];
}

extern "C" void kernel_launch(void* const* d_in, const int* in_sizes, int n_in,
                              void* d_out, int out_size, void* d_ws, size_t ws_size,
                              hipStream_t stream) {
  (void)in_sizes; (void)n_in; (void)out_size;
  const float* q = (const float*)d_in[0];
  const float* k = (const float*)d_in[1];
  const float* v = (const float*)d_in[2];
  const unsigned char* m = (const unsigned char*)d_in[3];  // jax bool = 1 byte
  float* out = (float*)d_out;
  dim3 grid(S_LEN / BLK_Q, NBH);

  const size_t kv_elems = (size_t)NBH * S_LEN * DHEAD;     // per tensor
  const size_t need = 2 * kv_elems * sizeof(uint16_t);     // 16.8 MB
  if (ws_size >= need) {
    __bf16* Kb  = (__bf16*)d_ws;
    __bf16* VTb = Kb + kv_elems;
    cvt_kv_kernel<<<dim3(NBH * 32), dim3(256), 0, stream>>>(k, v, Kb, VTb);
    fa_async_kernel<<<grid, dim3(256), 0, stream>>>(q, Kb, VTb, m, out);
  } else {
    fa_fallback_kernel<<<grid, dim3(256), 0, stream>>>(q, k, v, m, out);
  }
}
